// SelfAttention_82059645157872
// MI455X (gfx1250) — matrix operand order repaired and checked by reading, not compiled, not run on hardware
//
#include <hip/hip_runtime.h>
#include <hip/hip_bf16.h>

// Self-attention forward for MI455X (gfx1250, wave32, WMMA + TDM).
// GEMMs: v_wmma_f32_16x16x32_bf16 with fp32 accumulate; operand tiles staged
// into LDS by the Tensor Data Mover (tensor_load_to_lds, TENSORcnt tracked),
// double buffered.

#define B_  2
#define T_  2048
#define D_  1024
#define H_  16
#define DH_ 64

typedef __bf16 bf16_t;
typedef __bf16 v4bf  __attribute__((ext_vector_type(4)));
typedef __bf16 v8bf  __attribute__((ext_vector_type(8)));
typedef __bf16 v16bf __attribute__((ext_vector_type(16)));
typedef float  v4f   __attribute__((ext_vector_type(4)));
typedef float  v8f   __attribute__((ext_vector_type(8)));
typedef unsigned int u32x4 __attribute__((ext_vector_type(4)));
typedef int          i32x4 __attribute__((ext_vector_type(4)));
typedef int          i32x8 __attribute__((ext_vector_type(8)));

__device__ __forceinline__ v8f wmma_bf16(v16bf a, v16bf b, v8f c) {
  return __builtin_amdgcn_wmma_f32_16x16x32_bf16(false, a, false, b, (short)0, c,
                                                 false, false);
}

__device__ __forceinline__ v16bf pack16(v8bf lo, v8bf hi) {
  return __builtin_shufflevector(lo, hi, 0, 1, 2, 3, 4, 5, 6, 7,
                                 8, 9, 10, 11, 12, 13, 14, 15);
}

// ---------------- bf16-source fragment loaders (attention) ----------------
// A-fragment (16x32): lanes 0-15 row M=lane, K={0..7,16..23}; lanes 16-31 row
// M=lane-16, K={8..15,24..31}.
template <typename P>
__device__ __forceinline__ v16bf load_a_frag(const P* base, int row0, int stride,
                                             int kbase) {
  const int lane = threadIdx.x & 31;
  const int r    = lane & 15;
  const int k0   = kbase + ((lane >> 4) << 3);
  const P* p = base + (row0 + r) * stride + k0;
  v8bf lo = *(const v8bf*)(p);
  v8bf hi = *(const v8bf*)(p + 16);
  return pack16(lo, hi);
}

// B-fragment (32x16) from K-major "Bt" rows: lanes 0-15 col N=lane K={0..7,16..23},
// lanes 16-31 K={8..15,24..31}.
template <typename P>
__device__ __forceinline__ v16bf load_b_frag(const P* base, int col0, int stride,
                                             int kbase) {
  const int lane = threadIdx.x & 31;
  const int n    = lane & 15;
  const int kb   = kbase + ((lane >> 4) << 3);
  const P* p = base + (col0 + n) * stride + kb;
  v8bf lo = *(const v8bf*)(p);
  v8bf hi = *(const v8bf*)(p + 16);
  return pack16(lo, hi);
}

// ------------- fp32-in-LDS fragment loaders (TDM-staged GEMM) -------------
// LDS row stride = 40 floats (TDM hardware pad: 8 DWORDs per 32-DWORD row).
__device__ __forceinline__ v16bf a_frag_f32lds(const float* base, int row0) {
  const int lane = threadIdx.x & 31;
  const int r    = lane & 15;
  const int k0   = (lane >> 4) << 3;
  const float* p = base + (row0 + r) * 40 + k0;
  v8f lo = *(const v8f*)(p);
  v8f hi = *(const v8f*)(p + 16);
  return pack16(__builtin_convertvector(lo, v8bf), __builtin_convertvector(hi, v8bf));
}

__device__ __forceinline__ v16bf b_frag_f32lds(const float* base, int col0) {
  const int lane = threadIdx.x & 31;
  const int n    = lane & 15;
  const int kb   = (lane >> 4) << 3;
  const float* p = base + (col0 + n) * 40 + kb;
  v8f lo = *(const v8f*)(p);
  v8f hi = *(const v8f*)(p + 16);
  return pack16(__builtin_convertvector(lo, v8bf), __builtin_convertvector(hi, v8bf));
}

// ----------------------------- TDM tile load ------------------------------
// DMA a 128-row x 32-col fp32 tile (row-major, row stride = row_stride elems)
// into LDS with 8-DWORD padding after each 32-DWORD row (LDS stride 160 B).
// D# per CDNA5 ISA 8.3/8.4; groups 2/3 zero (2D tensor).
__device__ __forceinline__ void tdm_tile_load(const float* gtile,
                                              const float* lds_dst,
                                              int tensor_w, int row_stride) {
  const unsigned long long ga = (unsigned long long)gtile;
  const unsigned lds = (unsigned)(unsigned long long)lds_dst;  // LDS byte offset
  u32x4 g0 = {0u, 0u, 0u, 0u};
  g0[0] = 1u;                                            // count=1 (valid), user
  g0[1] = lds;                                           // lds_addr
  g0[2] = (unsigned)ga;                                  // global_addr[31:0]
  g0[3] = (unsigned)((ga >> 32) & 0x01ffffffu) | (2u << 30);  // ga[56:32], type=2
  i32x8 g1 = {0, 0, 0, 0, 0, 0, 0, 0};
  // data_size=2 (4B) | pad_enable | pad_interval code 4 (32 DW) | pad_amount code 7 (8 DW)
  g1[0] = (int)((2u << 16) | (1u << 20) | (4u << 22) | (7u << 25));
  g1[1] = (int)(((unsigned)tensor_w & 0xffffu) << 16);   // tensor_dim0 lo16
  g1[2] = (int)((((unsigned)tensor_w >> 16) & 0xffffu) | (128u << 16)); // dim0 hi | tensor_dim1=128
  g1[3] = (int)(32u << 16);                              // tensor_dim1 hi=0 | tile_dim0=32
  g1[4] = 128;                                           // tile_dim1=128, tile_dim2=0
  g1[5] = row_stride;                                    // tensor_dim0_stride[31:0]
  i32x4 z4 = {0, 0, 0, 0};
  i32x8 z8 = {0, 0, 0, 0, 0, 0, 0, 0};
  // 6-arg toolchain form: (g0, g1, g2, g3, extra, cpol)
  __builtin_amdgcn_tensor_load_to_lds(g0, g1, z4, z4, z8, 0);
}

// C = A * Bw^T (+ bias). A:[M,K] fp32, Bw:[N,K] fp32, 128x128x32 tiles,
// 8 waves (2x4), TDM double-buffered operand staging.
template <bool OUT_BF16>
__global__ void __launch_bounds__(256)
gemm_bf16wmma_kernel(const float* __restrict__ A, const float* __restrict__ Bw,
                     const float* __restrict__ bias, void* __restrict__ Cout,
                     int M, int N, int K) {
  __shared__ float As[2][128 * 40];   // 20 KiB per buffer
  __shared__ float Bs[2][128 * 40];

  const int tid  = threadIdx.x;
  const int w    = tid >> 5;
  const int wm   = w >> 2;  // 0..1 -> 64 M rows
  const int wn   = w & 3;   // 0..3 -> 32 N cols
  const int lane = tid & 31;
  const int hi   = lane >> 4;
  const int nn   = lane & 15;

  v8f acc[4][2];
#pragma unroll
  for (int mt = 0; mt < 4; ++mt)
#pragma unroll
    for (int nt = 0; nt < 2; ++nt)
#pragma unroll
      for (int r = 0; r < 8; ++r) acc[mt][nt][r] = 0.f;

  const float* gA = A + (size_t)(blockIdx.x * 128) * K;
  const float* gB = Bw + (size_t)(blockIdx.y * 128) * K;
  const int nk = K >> 5;

  if (w == 0) {
    tdm_tile_load(gA, As[0], K, K);
    tdm_tile_load(gB, Bs[0], K, K);
  }

  for (int it = 0; it < nk; ++it) {
    const int buf = it & 1;
    if (w == 0) __builtin_amdgcn_s_wait_tensorcnt(0);
    __syncthreads();  // current buffer's TDM data visible to all waves
    if (w == 0 && it + 1 < nk) {
      const int kb = (it + 1) << 5;
      tdm_tile_load(gA + kb, As[buf ^ 1], K, K);
      tdm_tile_load(gB + kb, Bs[buf ^ 1], K, K);
    }

    v16bf af[4], bfr[2];
#pragma unroll
    for (int mt = 0; mt < 4; ++mt)
      af[mt] = a_frag_f32lds(As[buf], wm * 64 + mt * 16);
#pragma unroll
    for (int nt = 0; nt < 2; ++nt)
      bfr[nt] = b_frag_f32lds(Bs[buf], wn * 32 + nt * 16);
#pragma unroll
    for (int mt = 0; mt < 4; ++mt)
#pragma unroll
      for (int nt = 0; nt < 2; ++nt)
        acc[mt][nt] = wmma_bf16(af[mt], bfr[nt], acc[mt][nt]);
    // no second barrier: the TDM that overwrites this buffer is issued only
    // after the *next* top-of-loop barrier, which all waves reach only after
    // finishing these LDS reads.
  }

#pragma unroll
  for (int mt = 0; mt < 4; ++mt) {
#pragma unroll
    for (int nt = 0; nt < 2; ++nt) {
#pragma unroll
      for (int r = 0; r < 8; ++r) {
        const int row = blockIdx.x * 128 + wm * 64 + mt * 16 + r + 8 * hi;
        const int col = blockIdx.y * 128 + wn * 32 + nt * 16 + nn;
        const float v = acc[mt][nt][r];
        if (OUT_BF16) {
          ((bf16_t*)Cout)[(size_t)row * N + col] = (bf16_t)v;
        } else {
          const float bv = bias ? bias[col] : 0.f;
          ((float*)Cout)[(size_t)row * N + col] = v + bv;
        }
      }
    }
  }
}

// vt[b,h,d,t] = V[b,t,h,d]; LDS tile transpose, 32B-coalesced both directions.
__global__ void __launch_bounds__(256)
vtrans_kernel(const bf16_t* __restrict__ qkv, bf16_t* __restrict__ vt) {
  __shared__ bf16_t tile[64][72];  // [t][d], +8 pad
  const int bh = blockIdx.x;
  const int t0 = blockIdx.y * 64;
  const int b  = bh >> 4;
  const int h  = bh & 15;
  const int tid = threadIdx.x;
  {
    const int r = tid >> 2;            // t row 0..63
    const int c = (tid & 3) << 4;      // d chunk
    const bf16_t* src =
        qkv + (size_t)(b * T_ + t0 + r) * (3 * D_) + 2 * D_ + h * DH_ + c;
    *(v8bf*)(&tile[r][c])     = *(const v8bf*)(src);
    *(v8bf*)(&tile[r][c + 8]) = *(const v8bf*)(src + 8);
  }
  __syncthreads();
  {
    const int d  = tid >> 2;           // dh row 0..63
    const int tc = (tid & 3) << 4;     // t chunk
    v8bf y0, y1;
#pragma unroll
    for (int i = 0; i < 8; ++i) { y0[i] = tile[tc + i][d]; y1[i] = tile[tc + 8 + i][d]; }
    bf16_t* dst = vt + (size_t)bh * DH_ * T_ + (size_t)d * T_ + t0 + tc;
    *(v8bf*)(dst)     = y0;
    *(v8bf*)(dst + 8) = y1;
  }
}

// Flash attention: block = (b*h, 64-row tile), 4 waves, wave owns 16 Q rows.
// Per-wave LDS scratch only -> no barriers.
__global__ void __launch_bounds__(128)
attn_fwd_kernel(const bf16_t* __restrict__ qkv, const bf16_t* __restrict__ vt,
                float* __restrict__ attn_out) {
  __shared__ bf16_t psh[4][16 * 72];

  const int bh   = blockIdx.x;
  const int b    = bh >> 4;
  const int h    = bh & 15;
  const int w    = threadIdx.x >> 5;
  const int lane = threadIdx.x & 31;
  const int hi   = lane >> 4;
  const int nn   = lane & 15;
  const int i0   = blockIdx.y * 64 + w * 16;

  const bf16_t* qbase = qkv + (size_t)b * T_ * (3 * D_) + h * DH_;
  const bf16_t* kbase = qbase + D_;
  const bf16_t* vtb   = vt + (size_t)bh * DH_ * T_;

  const v16bf qa0 = load_a_frag(qbase + (size_t)i0 * (3 * D_), 0, 3 * D_, 0);
  const v16bf qa1 = load_a_frag(qbase + (size_t)i0 * (3 * D_), 0, 3 * D_, 32);

  float m[8], lsum[8];
  v8f o[4];
#pragma unroll
  for (int r = 0; r < 8; ++r) { m[r] = -1e30f; lsum[r] = 0.f; }
#pragma unroll
  for (int nt = 0; nt < 4; ++nt)
#pragma unroll
    for (int r = 0; r < 8; ++r) o[nt][r] = 0.f;

  bf16_t* pw = psh[w];

  for (int j0 = 0; j0 <= i0; j0 += 64) {
    v8f s[4];
#pragma unroll
    for (int jt = 0; jt < 4; ++jt) {
      v8f c;
#pragma unroll
      for (int r = 0; r < 8; ++r) c[r] = 0.f;
      v16bf bk0 = load_b_frag(kbase, j0 + jt * 16, 3 * D_, 0);
      v16bf bk1 = load_b_frag(kbase, j0 + jt * 16, 3 * D_, 32);
      c = wmma_bf16(qa0, bk0, c);
      c = wmma_bf16(qa1, bk1, c);
      s[jt] = c;
    }

#pragma unroll
    for (int jt = 0; jt < 4; ++jt) {
#pragma unroll
      for (int r = 0; r < 8; ++r) {
        const int row = i0 + r + 8 * hi;
        const int col = j0 + jt * 16 + nn;
        const float v = s[jt][r] * 0.125f;
        s[jt][r] = (col > row) ? -1e30f : v;
      }
    }

#pragma unroll
    for (int r = 0; r < 8; ++r) {
      float mx = fmaxf(fmaxf(s[0][r], s[1][r]), fmaxf(s[2][r], s[3][r]));
#pragma unroll
      for (int off = 1; off < 16; off <<= 1) mx = fmaxf(mx, __shfl_xor(mx, off, 32));
      const float mnew  = fmaxf(m[r], mx);
      const float alpha = __expf(m[r] - mnew);
      m[r] = mnew;
      lsum[r] *= alpha;
#pragma unroll
      for (int nt = 0; nt < 4; ++nt) o[nt][r] *= alpha;
      float rs = 0.f;
#pragma unroll
      for (int jt = 0; jt < 4; ++jt) {
        const float p = __expf(s[jt][r] - mnew);
        s[jt][r] = p;
        rs += p;
      }
#pragma unroll
      for (int off = 1; off < 16; off <<= 1) rs += __shfl_xor(rs, off, 32);
      lsum[r] += rs;
    }

#pragma unroll
    for (int jt = 0; jt < 4; ++jt)
#pragma unroll
      for (int r = 0; r < 8; ++r)
        pw[(r + 8 * hi) * 72 + jt * 16 + nn] = (bf16_t)s[jt][r];

    const v16bf pa0 = load_a_frag((const bf16_t*)pw, 0, 72, 0);
    const v16bf pa1 = load_a_frag((const bf16_t*)pw, 0, 72, 32);

#pragma unroll
    for (int nt = 0; nt < 4; ++nt) {
      v16bf vb0 = load_b_frag(vtb, nt * 16, T_, j0);
      v16bf vb1 = load_b_frag(vtb, nt * 16, T_, j0 + 32);
      o[nt] = wmma_bf16(pa0, vb0, o[nt]);
      o[nt] = wmma_bf16(pa1, vb1, o[nt]);
    }
  }

#pragma unroll
  for (int r = 0; r < 8; ++r) {
    const float inv = 1.f / lsum[r];
    const int trow = i0 + r + 8 * hi;
#pragma unroll
    for (int nt = 0; nt < 4; ++nt) {
      const int col = h * DH_ + nt * 16 + nn;
      attn_out[((size_t)(b * T_ + trow)) * D_ + col] = o[nt][r] * inv;
    }
  }
}

extern "C" void kernel_launch(void* const* d_in, const int* in_sizes, int n_in,
                              void* d_out, int out_size, void* d_ws, size_t ws_size,
                              hipStream_t stream) {
  (void)in_sizes; (void)n_in; (void)out_size; (void)ws_size;
  const float* x    = (const float*)d_in[0];
  const float* Wqkv = (const float*)d_in[1];
  const float* Wout = (const float*)d_in[2];
  const float* bout = (const float*)d_in[3];
  float* out = (float*)d_out;

  char* ws = (char*)d_ws;  // needs ~50 MiB
  bf16_t* qkv  = (bf16_t*)(ws);                              // 4096x3072 bf16 = 24 MiB
  bf16_t* vt   = (bf16_t*)(ws + (size_t)25 * 1024 * 1024);   // [B*H][64][T] bf16 = 8 MiB
  float*  attn = (float*)(ws + (size_t)34 * 1024 * 1024);    // 4096x1024 fp32 = 16 MiB

  dim3 g1((B_ * T_) / 128, (3 * D_) / 128);
  gemm_bf16wmma_kernel<true><<<g1, 256, 0, stream>>>(x, Wqkv, nullptr, qkv,
                                                     B_ * T_, 3 * D_, D_);
  dim3 gv(B_ * H_, T_ / 64);
  vtrans_kernel<<<gv, 256, 0, stream>>>(qkv, vt);
  dim3 g2(B_ * H_, T_ / 64);
  attn_fwd_kernel<<<g2, 128, 0, stream>>>(qkv, vt, attn);
  dim3 g3((B_ * T_) / 128, D_ / 128);
  gemm_bf16wmma_kernel<false><<<g3, 256, 0, stream>>>(attn, Wout, bout, out,
                                                      B_ * T_, D_, D_);
}
